// GeneExpressionGNN_52879637348574
// MI455X (gfx1250) — matrix operand order repaired and checked by reading, not compiled, not run on hardware
//
#include <hip/hip_runtime.h>
#include <hip/hip_bf16.h>

typedef __attribute__((ext_vector_type(16))) __bf16 v16bf;
typedef __attribute__((ext_vector_type(8)))  __bf16 v8bf;
typedef __attribute__((ext_vector_type(8)))  float  v8f;

#define TM 64
#define TN 128
#define TK 32
#define LPAD 8   // 16B pad per LDS row

// ---------------------------------------------------------------------------
// Utility kernels
// ---------------------------------------------------------------------------
__global__ void fill_f32(float* __restrict__ p, float v, int n) {
    int i = blockIdx.x * blockDim.x + threadIdx.x;
    if (i < n) p[i] = v;
}

__global__ void cvt_f32_bf16(const float* __restrict__ in, __bf16* __restrict__ out, int n) {
    int i = blockIdx.x * blockDim.x + threadIdx.x;
    if (i < n) out[i] = (__bf16)in[i];
}

// Wt[n*K + k] = bf16(W[k*N + n])  (transposed bf16 weights, K-contiguous rows)
__global__ void cvt_transpose_bf16(const float* __restrict__ W, __bf16* __restrict__ Wt,
                                   int K, int N) {
    int i = blockIdx.x * blockDim.x + threadIdx.x;
    if (i < K * N) {
        int k = i / N, n = i % N;
        Wt[(size_t)n * K + k] = (__bf16)W[i];
    }
}

__global__ void deg_edges(const int* __restrict__ dst, float* __restrict__ deg, int E) {
    int i = blockIdx.x * blockDim.x + threadIdx.x;
    if (i < E) atomicAdd(&deg[dst[i]], 1.0f);
}

__global__ void inv_sqrt_deg_k(const float* __restrict__ deg, float* __restrict__ isd, int n) {
    int i = blockIdx.x * blockDim.x + threadIdx.x;
    if (i < n) {
        float d = deg[i];                 // >= 1 always (self-loop)
        isd[i] = rsqrtf(d < 1.0f ? 1.0f : d);
    }
}

// ---------------------------------------------------------------------------
// bf16 WMMA GEMM: C[M,N] = A[M,K](bf16, row-major) * Bt[N,K](bf16, transposed),
// f32 accumulate. 256 threads = 8 waves; tile 64x128; wave -> 16x64 patch.
// Double-buffered LDS; all fragment traffic is ds_load_b128.
// ---------------------------------------------------------------------------
template <int K>
__global__ __launch_bounds__(256)
void gemm_bf16_wmma(const __bf16* __restrict__ A, const __bf16* __restrict__ Bt,
                    float* __restrict__ C, int M, int N) {
    __shared__ __bf16 sA [2][TM][TK + LPAD];
    __shared__ __bf16 sBt[2][TN][TK + LPAD];

    const int tid   = threadIdx.x;
    const int lane  = tid & 31;
    const int wv    = tid >> 5;         // 0..7
    const int m_sub = wv & 3;           // 16-row sub-tile
    const int n_sub = wv >> 2;          // 0..1 -> 64-col group
    const int mbase = blockIdx.y * TM;
    const int nbase = blockIdx.x * TN;

    v8f acc[4] = {};

    // A loader: thread -> (row 0..63, 8-elem K segment)
    const int arow = tid >> 2;
    const int aseg = (tid & 3) * 8;
    int agr = mbase + arow; if (agr > M - 1) agr = M - 1;   // clamp; stores guarded
    const __bf16* aptr = A + (size_t)agr * K + aseg;

    // B loader: thread j-th pass -> (col 0..127, 8-elem K segment)
    const int bcol0 = tid >> 1;               // j=0: cols 0..127 (two segs/thread)
    const int bseg0 = (tid & 1) * 16;

    auto load_tile = [&](int buf, int kk) {
        __builtin_prefetch(aptr + kk + TK, 0, 0);           // global_prefetch_b8
        *(uint4*)&sA[buf][arow][aseg] = *(const uint4*)(aptr + kk);
#pragma unroll
        for (int j = 0; j < 2; ++j) {
            const int idx = tid + j * 256;                  // 0..511
            const int col = idx >> 2;                       // 0..127
            const int seg = (idx & 3) * 8;                  // 0,8,16,24
            *(uint4*)&sBt[buf][col][seg] =
                *(const uint4*)(Bt + (size_t)(nbase + col) * K + kk + seg);
        }
    };

    load_tile(0, 0);
    __syncthreads();

#pragma unroll
    for (int kk = 0; kk < K; kk += TK) {
        const int buf = (kk / TK) & 1;
        if (kk + TK < K) load_tile(buf ^ 1, kk + TK);

        // A fragment (16-bit A 16x32 layout): lane holds row lane%16;
        // elems 0..7: K=(lane/16)*8+e ; elems 8..15: K=16+(lane/16)*8+e
        const __bf16* ap = &sA[buf][m_sub * 16 + (lane & 15)][0];
        const int kh = (lane >> 4) * 8;
        v8bf alo = *(const v8bf*)(ap + kh);
        v8bf ahi = *(const v8bf*)(ap + 16 + kh);
        v16bf afrag = __builtin_shufflevector(alo, ahi,
            0, 1, 2, 3, 4, 5, 6, 7, 8, 9, 10, 11, 12, 13, 14, 15);

        // B fragments: lane holds col lane%16; elems 0..7: K=(lane/16)*8+e ; elems 8..15: K=16+(lane/16)*8+e
#pragma unroll
        for (int t = 0; t < 4; ++t) {
            const __bf16* bp =
                &sBt[buf][n_sub * 64 + t * 16 + (lane & 15)][(lane >> 4) * 8];
            v8bf blo = *(const v8bf*)(bp);
            v8bf bhi = *(const v8bf*)(bp + 16);
            v16bf bfrag = __builtin_shufflevector(blo, bhi,
                0, 1, 2, 3, 4, 5, 6, 7, 8, 9, 10, 11, 12, 13, 14, 15);
            acc[t] = __builtin_amdgcn_wmma_f32_16x16x32_bf16(
                false, afrag, false, bfrag, (short)0, acc[t], false, false);
        }
        __syncthreads();
    }

    // C/D layout: lane holds col lane%16; VGPR r -> row (lane/16)*8 + r
#pragma unroll
    for (int t = 0; t < 4; ++t) {
        const int col = nbase + n_sub * 64 + t * 16 + (lane & 15);
#pragma unroll
        for (int r = 0; r < 8; ++r) {
            const int row = mbase + m_sub * 16 + (lane >> 4) * 8 + r;
            if (row < M) C[(size_t)row * N + col] = acc[t][r];
        }
    }
}

// ---------------------------------------------------------------------------
// Graph aggregation
// ---------------------------------------------------------------------------
__global__ void agg_init(const float* __restrict__ hw, const float* __restrict__ isd,
                         float* __restrict__ agg, int total) {
    int i = blockIdx.x * blockDim.x + threadIdx.x;
    if (i < total) {
        float s = isd[i >> 8];
        agg[i] = hw[i] * s * s;           // self-loop term
    }
}

// one wave32 per edge: float4 gather of 1KB row, 256 f32 atomics scatter
__global__ __launch_bounds__(256)
void scatter_edges(const float* __restrict__ hw, const int* __restrict__ src,
                   const int* __restrict__ dst, const float* __restrict__ isd,
                   float* __restrict__ agg, int E) {
    const int e    = (int)((blockIdx.x * (size_t)blockDim.x + threadIdx.x) >> 5);
    const int lane = threadIdx.x & 31;
    if (e >= E) return;
    const int s = src[e], d = dst[e];
    const float norm = isd[s] * isd[d];
    const float4* hs = (const float4*)(hw + (size_t)s * 256);
    float* ad = agg + (size_t)d * 256;
#pragma unroll
    for (int j = 0; j < 2; ++j) {
        const int q = lane + j * 32;
        float4 v = hs[q];
        atomicAdd(&ad[q * 4 + 0], v.x * norm);
        atomicAdd(&ad[q * 4 + 1], v.y * norm);
        atomicAdd(&ad[q * 4 + 2], v.z * norm);
        atomicAdd(&ad[q * 4 + 3], v.w * norm);
    }
}

__global__ void bias_relu_cvt(float* __restrict__ agg, const float* __restrict__ bias,
                              __bf16* __restrict__ hbf, int total) {
    int i = blockIdx.x * blockDim.x + threadIdx.x;
    if (i < total) {
        float v = agg[i] + bias[i & 255];
        v = v > 0.0f ? v : 0.0f;
        agg[i] = v;                        // f32 kept for pooling
        hbf[i] = (__bf16)v;                // bf16 for next GEMM
    }
}

// ---------------------------------------------------------------------------
// Pooling + FC head (64 graphs -> tiny, latency-bound)
// ---------------------------------------------------------------------------
__global__ void pool_count(const int* __restrict__ batch, float* __restrict__ cnt, int N) {
    int i = blockIdx.x * blockDim.x + threadIdx.x;
    if (i < N) atomicAdd(&cnt[batch[i]], 1.0f);
}

__global__ void pool_sum(const float* __restrict__ h, const int* __restrict__ batch,
                         float* __restrict__ psum, int total) {
    int i = blockIdx.x * blockDim.x + threadIdx.x;
    if (i < total) atomicAdd(&psum[(size_t)batch[i >> 8] * 256 + (i & 255)], h[i]);
}

__global__ void pool_div(float* __restrict__ psum, const float* __restrict__ cnt, int total) {
    int i = blockIdx.x * blockDim.x + threadIdx.x;
    if (i < total) {
        float c = cnt[i >> 8];
        psum[i] /= (c < 1.0f ? 1.0f : c);
    }
}

__global__ void fc_relu(const float* __restrict__ in, const float* __restrict__ W,
                        const float* __restrict__ b, float* __restrict__ out,
                        int Din, int Dout, int total) {
    int i = blockIdx.x * blockDim.x + threadIdx.x;
    if (i >= total) return;
    int g = i / Dout, j = i % Dout;
    const float* row = in + (size_t)g * Din;
    float acc = b[j];
    for (int k = 0; k < Din; ++k) acc += row[k] * W[(size_t)k * Dout + j];
    out[i] = acc > 0.0f ? acc : 0.0f;
}

__global__ void fc_lin(const float* __restrict__ in, const float* __restrict__ W,
                       const float* __restrict__ b, float* __restrict__ out,
                       int Din, int Dout, int total) {
    int i = blockIdx.x * blockDim.x + threadIdx.x;
    if (i >= total) return;
    int g = i / Dout, j = i % Dout;
    const float* row = in + (size_t)g * Din;
    float acc = b[j];
    for (int k = 0; k < Din; ++k) acc += row[k] * W[(size_t)k * Dout + j];
    out[i] = acc;
}

// ---------------------------------------------------------------------------
extern "C" void kernel_launch(void* const* d_in, const int* in_sizes, int n_in,
                              void* d_out, int out_size, void* d_ws, size_t ws_size,
                              hipStream_t stream) {
    (void)n_in; (void)out_size; (void)ws_size;
    const float* x    = (const float*)d_in[0];
    const int*   ei   = (const int*)d_in[1];
    const int*   bat  = (const int*)d_in[2];
    const float* Wg[3]   = {(const float*)d_in[3], (const float*)d_in[5], (const float*)d_in[7]};
    const float* bias[3] = {(const float*)d_in[4], (const float*)d_in[6], (const float*)d_in[8]};
    const float* fc1w = (const float*)d_in[9];
    const float* fc1b = (const float*)d_in[10];
    const float* fc2w = (const float*)d_in[11];
    const float* fc2b = (const float*)d_in[12];

    const int D = 256, Dout = 128, G = 64;
    const int N = in_sizes[0] / D;
    const int E = in_sizes[1] / 2;
    const int* src = ei;
    const int* dst = ei + E;

    char* base = (char*)d_ws; size_t off = 0;
    auto carve = [&](size_t bytes) -> void* {
        void* p = base + off; off = (off + bytes + 255) & ~(size_t)255; return p;
    };
    float*  deg  = (float*) carve((size_t)N * 4);
    float*  isd  = (float*) carve((size_t)N * 4);
    __bf16* hbf  = (__bf16*)carve((size_t)N * D * 2);
    float*  hw   = (float*) carve((size_t)N * D * 4);
    float*  agg  = (float*) carve((size_t)N * D * 4);
    __bf16* wtb[3];
    for (int l = 0; l < 3; ++l) wtb[l] = (__bf16*)carve((size_t)D * D * 2);
    float*  psum = (float*) carve((size_t)G * D * 4);
    float*  pcnt = (float*) carve((size_t)G * 4);
    float*  fc1o = (float*) carve((size_t)G * D * 4);

    const int BT = 256;
    const int ND = N * D;

    // degrees (self-loop counted via init=1.0) and D^{-1/2}
    fill_f32<<<(N + BT - 1) / BT, BT, 0, stream>>>(deg, 1.0f, N);
    deg_edges<<<(E + BT - 1) / BT, BT, 0, stream>>>(dst, deg, E);
    inv_sqrt_deg_k<<<(N + BT - 1) / BT, BT, 0, stream>>>(deg, isd, N);

    // bf16 operands: activations row-major, weights transposed (N-major, K-contig)
    cvt_f32_bf16<<<(ND + BT - 1) / BT, BT, 0, stream>>>(x, hbf, ND);
    for (int l = 0; l < 3; ++l)
        cvt_transpose_bf16<<<(D * D + BT - 1) / BT, BT, 0, stream>>>(Wg[l], wtb[l], D, D);

    dim3 ggrid(D / TN, (N + TM - 1) / TM);
    size_t sblk = ((size_t)E * 32 + BT - 1) / BT;
    for (int l = 0; l < 3; ++l) {
        gemm_bf16_wmma<256><<<ggrid, BT, 0, stream>>>(hbf, wtb[l], hw, N, D);
        agg_init<<<(ND + BT - 1) / BT, BT, 0, stream>>>(hw, isd, agg, ND);
        scatter_edges<<<(unsigned)sblk, BT, 0, stream>>>(hw, src, dst, isd, agg, E);
        bias_relu_cvt<<<(ND + BT - 1) / BT, BT, 0, stream>>>(agg, bias[l], hbf, ND);
    }

    // global mean pool
    fill_f32<<<(G * D + BT - 1) / BT, BT, 0, stream>>>(psum, 0.0f, G * D);
    fill_f32<<<1, G, 0, stream>>>(pcnt, 0.0f, G);
    pool_count<<<(N + BT - 1) / BT, BT, 0, stream>>>(bat, pcnt, N);
    pool_sum<<<(ND + BT - 1) / BT, BT, 0, stream>>>(agg, bat, psum, ND);
    pool_div<<<(G * D + BT - 1) / BT, BT, 0, stream>>>(psum, pcnt, G * D);

    // FC head
    fc_relu<<<(G * D + BT - 1) / BT, BT, 0, stream>>>(psum, fc1w, fc1b, fc1o, D, D, G * D);
    fc_lin<<<(G * Dout + BT - 1) / BT, BT, 0, stream>>>(fc1o, fc2w, fc2b, (float*)d_out,
                                                        D, Dout, G * Dout);
}
